// LearnableSparseAttention_33389075759543
// MI455X (gfx1250) — matrix operand order repaired and checked by reading, not compiled, not run on hardware
//
#include <hip/hip_runtime.h>
#include <hip/hip_bf16.h>
#include <math.h>

// ---- problem dims ----
#define Bb   8
#define Nn   1024
#define Cc   512
#define Hh   8
#define BDm  6
#define HDd  64
// ---- attention tiling ----
#define ROWS_WG 32          // 2 waves x 16 rows
#define SC_PITCH 1032       // padded score-row pitch (floats)
#define FRAG     512        // bf16 elems per 16x16x32 fragment (32 lanes x 16)

typedef __attribute__((ext_vector_type(16))) __bf16 v16bf;
typedef __attribute__((ext_vector_type(8)))  float  v8f;

static __device__ __forceinline__ v16bf ldfrag_g(const __bf16* p) { return *(const v16bf*)p; }
static __device__ __forceinline__ v16bf ldfrag_l(const __bf16* p) { return *(const v16bf*)p; }

// =====================================================================
// Kernel 0a: pack adjacency into a bit matrix (128 KB, L2-resident)
// =====================================================================
__global__ void adjbits_kernel(const int* __restrict__ adj,
                               unsigned int* __restrict__ adjb)
{
    const int n = blockIdx.x;
    const int w = threadIdx.x;          // 32 words per row
    unsigned int bits = 0u;
    #pragma unroll
    for (int j = 0; j < 32; ++j)
        bits |= (adj[(size_t)n * Nn + w * 32 + j] > 0 ? 1u : 0u) << j;
    adjb[n * 32 + w] = bits;
}

// =====================================================================
// Kernel 1: LayerNorm + low-rank QKV projection.
// Writes q / k^T / v in WMMA *fragment-ready* bf16 layouts so the
// attention kernel loads each operand fragment as one 32B contiguous load.
//   qfrag: [BH][rowtile 64][kstep 2][lane 32][slot 16]   (A operand, q/8)
//   kfrag: [BH][ntile  64][kstep 2][lane 32][slot 16]    (B operand, k^T)
//   vfrag: [BH][kstep 32][dtile 4][lane 32][slot 16]     (B operand, v)
// one 256-thread block per (b,n) row
// =====================================================================
__global__ void ln_qkv_kernel(const float* __restrict__ x,
                              const float* __restrict__ ln_g,
                              const float* __restrict__ ln_b,
                              const float* __restrict__ w_low,
                              const float* __restrict__ b_low,
                              const float* __restrict__ w_high,
                              const float* __restrict__ b_high,
                              __bf16* __restrict__ qfrag,
                              __bf16* __restrict__ kfrag,
                              __bf16* __restrict__ vfrag)
{
    __shared__ float red[256];
    __shared__ float xn[Cc];
    __shared__ float t[3 * BDm];

    const int row = blockIdx.x;          // b*N + n
    const int b   = row / Nn;
    const int n   = row % Nn;
    const int tid = threadIdx.x;

    const float* xr = x + (size_t)row * Cc;
    const float x0 = xr[tid];
    const float x1 = xr[tid + 256];

    red[tid] = x0 + x1;
    __syncthreads();
    for (int s = 128; s > 0; s >>= 1) {
        if (tid < s) red[tid] += red[tid + s];
        __syncthreads();
    }
    const float mu = red[0] * (1.0f / Cc);
    __syncthreads();

    red[tid] = (x0 - mu) * (x0 - mu) + (x1 - mu) * (x1 - mu);
    __syncthreads();
    for (int s = 128; s > 0; s >>= 1) {
        if (tid < s) red[tid] += red[tid + s];
        __syncthreads();
    }
    const float rs = rsqrtf(red[0] * (1.0f / Cc) + 1e-5f);
    __syncthreads();

    xn[tid]       = (x0 - mu) * rs * ln_g[tid]       + ln_b[tid];
    xn[tid + 256] = (x1 - mu) * rs * ln_g[tid + 256] + ln_b[tid + 256];
    if (tid < 3 * BDm) t[tid] = b_low[tid];
    __syncthreads();

    const float v0 = xn[tid], v1 = xn[tid + 256];
    #pragma unroll
    for (int r = 0; r < 3 * BDm; ++r)
        atomicAdd(&t[r], v0 * w_low[tid * (3 * BDm) + r] +
                         v1 * w_low[(tid + 256) * (3 * BDm) + r]);
    __syncthreads();

    #pragma unroll
    for (int jj = 0; jj < 6; ++jj) {
        const int j = tid + jj * 256;
        float o = b_high[j];
        #pragma unroll
        for (int r = 0; r < 3 * BDm; ++r)
            o += t[r] * w_high[r * (3 * Cc) + j];
        const int s  = j / Cc;       // 0=q 1=k 2=v
        const int jc = j % Cc;
        const int h  = jc / HDd;
        const int d  = jc % HDd;
        const size_t bh = (size_t)(b * Hh + h);
        if (s == 0) {
            // A-frag layout: lane = M + 16*K[4]; slot = K[3:0]
            const int ks   = d >> 5;
            const int K    = d & 31;
            const int lane = (n & 15) + ((K >> 4) << 4);
            const int slot = K & 15;
            qfrag[((bh * 64 + (n >> 4)) * 2 + ks) * FRAG + lane * 16 + slot] =
                (__bf16)(o * 0.125f);                       // fold 1/sqrt(64)
        } else if (s == 1) {
            // B-frag layout (k^T): lane = N + 16*K[4]; slot = K[3:0]
            const int ks   = d >> 5;
            const int Kb   = d & 31;
            const int lane = (n & 15) + ((Kb >> 4) << 4);
            kfrag[((bh * 64 + (n >> 4)) * 2 + ks) * FRAG + lane * 16 + (Kb & 15)] =
                (__bf16)o;
        } else {
            // B-frag layout (v): K dim is the token index n
            const int ks   = n >> 5;
            const int Kb   = n & 31;
            const int lane = (d & 15) + ((Kb >> 4) << 4);
            vfrag[((bh * 32 + ks) * 4 + (d >> 4)) * FRAG + lane * 16 + (Kb & 15)] =
                (__bf16)o;
        }
    }
}

// =====================================================================
// Kernel 2: fused sparse attention.
// One 64-thread block (2 waves) per (b, h, 32-row block). Per wave: 16 rows.
//  phase 1: score strip 16x1024 via v_wmma_f32_16x16x32_bf16
//           (B frags = contiguous 32B global loads) + rank-6 bias + adj bits
//  phase 2: per-row top-k (value binary search on LDS strip) + softmax;
//           probabilities written as bf16 straight into a swizzled LDS
//           A-fragment buffer; reg-loss atomics
//  phase 3: attn @ V via WMMA (A = ds_load_b128 pairs, B = 32B global loads)
// =====================================================================
__global__ void attn_kernel(const __bf16* __restrict__ qfrag,
                            const __bf16* __restrict__ kfrag,
                            const __bf16* __restrict__ vfrag,
                            const unsigned int* __restrict__ adjb,
                            const float*  __restrict__ su,
                            const float*  __restrict__ sv,
                            const float*  __restrict__ adj_scale_p,
                            const float*  __restrict__ thr,
                            float*        __restrict__ att,
                            float*        __restrict__ accum)
{
    extern __shared__ float smem[];
    float*        sc     = smem;                                    // [32][SC_PITCH]
    float*        su_s   = sc + ROWS_WG * SC_PITCH;                 // [32][6]
    float*        sv_s   = su_s + ROWS_WG * BDm;                    // [6][1024]
    unsigned int* adjb_s = (unsigned int*)(sv_s + BDm * Nn);        // [32][32]
    __bf16*       afrag  = (__bf16*)(adjb_s + ROWS_WG * 32);        // [2][32][FRAG]

    const int bidx = blockIdx.x;
    const int rb   = bidx & 31;                  // 32-row block within (b,h)
    const int bh   = bidx >> 5;
    const int h    = bh & 7;
    const int tid  = threadIdx.x;
    const int wave = tid >> 5;
    const int lane = tid & 31;
    const int lrow = lane & 15;
    const int half = lane >> 4;
    const float ascale = adj_scale_p[0];

    // ---- stage su rows, sv (this head), adjacency bit rows ----
    for (int i = tid; i < ROWS_WG * BDm; i += 64)
        su_s[i] = su[((size_t)h * Nn + rb * ROWS_WG + i / BDm) * BDm + i % BDm];
    for (int i = tid; i < BDm * Nn; i += 64)
        sv_s[i] = sv[(size_t)h * BDm * Nn + i];
    for (int i = tid; i < ROWS_WG * 32; i += 64)
        adjb_s[i] = adjb[(size_t)(rb * ROWS_WG + (i >> 5)) * 32 + (i & 31)];
    __syncthreads();

    const int wrow = wave * 16;
    const int rt   = rb * 2 + wave;              // row-tile index within (b,h)

    // ---- A fragments (q rows) : two 32B contiguous loads ----
    const size_t qfb = ((size_t)bh * 64 + rt) * 2 * FRAG + lane * 16;
    const v16bf aq0 = ldfrag_g(qfrag + qfb);
    const v16bf aq1 = ldfrag_g(qfrag + qfb + FRAG);

    // -------- phase 1: score strip --------
    const size_t kfb = (size_t)bh * 64 * 2 * FRAG + lane * 16;
    unsigned int aw[8];
    for (int nt = 0; nt < 64; ++nt) {
        const v16bf bk0 = ldfrag_g(kfrag + kfb + (size_t)nt * 2 * FRAG);
        const v16bf bk1 = ldfrag_g(kfrag + kfb + (size_t)nt * 2 * FRAG + FRAG);
        if (nt + 1 < 64)
            __builtin_prefetch(kfrag + kfb + (size_t)(nt + 1) * 2 * FRAG, 0, 0);

        v8f acc = {};
        acc = __builtin_amdgcn_wmma_f32_16x16x32_bf16(false, aq0, false, bk0, (short)0, acc, false, false);
        acc = __builtin_amdgcn_wmma_f32_16x16x32_bf16(false, aq1, false, bk1, (short)0, acc, false, false);

        const int m = nt * 16 + lrow;
        if ((nt & 1) == 0) {                       // adjacency word covers 2 ntiles
            #pragma unroll
            for (int i = 0; i < 8; ++i)
                aw[i] = adjb_s[(wrow + i + half * 8) * 32 + (nt >> 1)];
        }
        float svm[BDm];
        #pragma unroll
        for (int r = 0; r < BDm; ++r) svm[r] = sv_s[r * Nn + m];
        #pragma unroll
        for (int i = 0; i < 8; ++i) {
            const int rloc = wrow + i + half * 8;   // C layout: M = i + 8*half
            float bias = 0.f;
            #pragma unroll
            for (int r = 0; r < BDm; ++r) bias += su_s[rloc * BDm + r] * svm[r];
            const float maskb = ((aw[i] >> (m & 31)) & 1u) ? 0.f : -1e9f * ascale;
            sc[rloc * SC_PITCH + m] = acc[i] + bias + maskb;
        }
    }
    __syncthreads();

    // -------- phase 2: per-row top-k + softmax -> bf16 A-frags in LDS --------
    const float frac = 1.f / (1.f + __expf(thr[h]));      // 1 - sigmoid(thr)
    int kval = (int)((float)Nn * frac);
    if (kval < 1) kval = 1;

    float l1_acc = 0.f, ent_acc = 0.f;
    for (int rr = 0; rr < 16; ++rr) {
        float* rowp = &sc[(wrow + rr) * SC_PITCH];
        float mx = -3.4e38f, mn = 3.4e38f;
        for (int j = 0; j < 32; ++j) {
            const float v = rowp[lane + j * 32];
            mx = fmaxf(mx, v); mn = fminf(mn, v);
        }
        for (int o = 16; o; o >>= 1) {
            mx = fmaxf(mx, __shfl_xor(mx, o, 32));
            mn = fminf(mn, __shfl_xor(mn, o, 32));
        }
        // value binary search for the k-th largest
        float lo = mn, hi3 = mx;
        for (int it = 0; it < 30; ++it) {
            const float mid = 0.5f * (lo + hi3);
            int cnt = 0;
            for (int j = 0; j < 32; ++j) cnt += (rowp[lane + j * 32] >= mid) ? 1 : 0;
            for (int o = 16; o; o >>= 1) cnt += __shfl_xor(cnt, o, 32);
            if (cnt >= kval) lo = mid; else hi3 = mid;
        }
        const float kth = lo;
        float sum = 0.f;
        for (int j = 0; j < 32; ++j) {
            const int idx = lane + j * 32;
            const float v = rowp[idx];
            const float e = (v >= kth) ? __expf(v - mx) : 0.f;
            rowp[idx] = e; sum += e;
        }
        for (int o = 16; o; o >>= 1) sum += __shfl_xor(sum, o, 32);
        const float inv = 1.f / sum;
        for (int j = 0; j < 32; ++j) {
            const int idx = lane + j * 32;
            const float p = rowp[idx] * inv;
            l1_acc  += p;
            ent_acc -= p * __logf(p + 1e-9f);
            // scatter p as bf16 into the phase-3 A-fragment layout:
            // ks = idx>>5 ; K = idx&31 ; lane' = rr + 16*K[4] ; slot = K[3:0]
            const int K    = idx & 31;
            const int lnp  = rr + ((K >> 4) << 4);
            const int slot = K & 15;
            afrag[(wave * 32 + (idx >> 5)) * FRAG + lnp * 16 + slot] = (__bf16)p;
        }
    }
    for (int o = 16; o; o >>= 1) {
        l1_acc  += __shfl_xor(l1_acc,  o, 32);
        ent_acc += __shfl_xor(ent_acc, o, 32);
    }
    if (lane == 0) {
        atomicAdd(&accum[0], l1_acc);
        atomicAdd(&accum[1], ent_acc);
    }
    __syncthreads();

    // -------- phase 3: attended = attn @ V --------
    const size_t vfb = (size_t)bh * 32 * 4 * FRAG + lane * 16;
    v8f ad[4] = {};
    for (int ks = 0; ks < 32; ++ks) {
        const v16bf ap = ldfrag_l(afrag + (wave * 32 + ks) * FRAG + lane * 16);
        #pragma unroll
        for (int dt = 0; dt < 4; ++dt) {
            const v16bf bv = ldfrag_g(vfrag + vfb + (size_t)(ks * 4 + dt) * FRAG);
            ad[dt] = __builtin_amdgcn_wmma_f32_16x16x32_bf16(false, ap, false, bv, (short)0, ad[dt], false, false);
        }
    }
    const int b = bh >> 3;
    #pragma unroll
    for (int dt = 0; dt < 4; ++dt) {
        #pragma unroll
        for (int i = 0; i < 8; ++i) {
            const int gn = rb * ROWS_WG + wrow + i + half * 8;
            att[((size_t)b * Nn + gn) * Cc + h * HDd + dt * 16 + lrow] = ad[dt][i];
        }
    }
}

// =====================================================================
// Kernel 3: out = relu(att @ w_o1 + b_o1) @ w_o2 + b_o2 + x
// =====================================================================
__global__ void out_proj_kernel(const float* __restrict__ att,
                                const float* __restrict__ x,
                                const float* __restrict__ w_o1,
                                const float* __restrict__ b_o1,
                                const float* __restrict__ w_o2,
                                const float* __restrict__ b_o2,
                                float* __restrict__ out)
{
    __shared__ float t[BDm];
    const int row = blockIdx.x;
    const int tid = threadIdx.x;
    if (tid < BDm) t[tid] = b_o1[tid];
    __syncthreads();

    const float* ar = att + (size_t)row * Cc;
    const float a0 = ar[tid], a1 = ar[tid + 256];
    #pragma unroll
    for (int j = 0; j < BDm; ++j)
        atomicAdd(&t[j], a0 * w_o1[tid * BDm + j] + a1 * w_o1[(tid + 256) * BDm + j]);
    __syncthreads();
    if (tid < BDm) t[tid] = fmaxf(t[tid], 0.f);
    __syncthreads();

    #pragma unroll
    for (int u = 0; u < 2; ++u) {
        const int c = tid + u * 256;
        float o = b_o2[c] + x[(size_t)row * Cc + c];
        #pragma unroll
        for (int j = 0; j < BDm; ++j) o += t[j] * w_o2[j * Cc + c];
        out[(size_t)row * Cc + c] = o;
    }
}

__global__ void init_accum_kernel(float* accum)
{
    if (threadIdx.x < 2) accum[threadIdx.x] = 0.f;
}

__global__ void finalize_kernel(const float* __restrict__ accum,
                                const float* __restrict__ l1_w,
                                const float* __restrict__ ent_w,
                                float* __restrict__ out_scalar)
{
    const float sp1  = log1pf(__expf(l1_w[0]));
    const float sp2  = log1pf(__expf(ent_w[0]));
    const float l1m  = accum[0] / ((float)Bb * Hh * Nn * Nn);
    const float entm = accum[1] / ((float)Bb * Hh * Nn);
    out_scalar[0] = sp1 * l1m - sp2 * entm;
}

// =====================================================================
extern "C" void kernel_launch(void* const* d_in, const int* in_sizes, int n_in,
                              void* d_out, int out_size, void* d_ws, size_t ws_size,
                              hipStream_t stream)
{
    (void)in_sizes; (void)n_in; (void)out_size; (void)ws_size;

    const float* x         = (const float*)d_in[0];
    const int*   adj       = (const int*)  d_in[1];
    const float* ln_g      = (const float*)d_in[2];
    const float* ln_b      = (const float*)d_in[3];
    const float* w_low     = (const float*)d_in[4];
    const float* b_low     = (const float*)d_in[5];
    const float* w_high    = (const float*)d_in[6];
    const float* b_high    = (const float*)d_in[7];
    const float* su        = (const float*)d_in[8];
    const float* sv        = (const float*)d_in[9];
    const float* adj_scale = (const float*)d_in[10];
    const float* thr       = (const float*)d_in[11];
    const float* l1_w      = (const float*)d_in[12];
    const float* ent_w     = (const float*)d_in[13];
    const float* w_o1      = (const float*)d_in[14];
    const float* b_o1      = (const float*)d_in[15];
    const float* w_o2      = (const float*)d_in[16];
    const float* b_o2      = (const float*)d_in[17];

    char* ws = (char*)d_ws;
    const size_t frag_elems = (size_t)Bb * Hh * Nn * HDd;        // 4M bf16 per tensor
    __bf16* qfrag = (__bf16*)(ws);                               // 8 MB
    __bf16* kfrag = (__bf16*)(ws + frag_elems * 2);              // 8 MB
    __bf16* vfrag = (__bf16*)(ws + frag_elems * 4);              // 8 MB
    float*  att   = (float*) (ws + frag_elems * 6);              // 16 MB
    unsigned int* adjbw = (unsigned int*)(ws + frag_elems * 6 + (size_t)Bb * Nn * Cc * 4);
    float*  accum = (float*)((char*)adjbw + (size_t)Nn * 32 * 4);

    float* out = (float*)d_out;

    init_accum_kernel<<<dim3(1), dim3(32), 0, stream>>>(accum);

    adjbits_kernel<<<dim3(Nn), dim3(32), 0, stream>>>(adj, adjbw);

    ln_qkv_kernel<<<dim3(Bb * Nn), dim3(256), 0, stream>>>(
        x, ln_g, ln_b, w_low, b_low, w_high, b_high, qfrag, kfrag, vfrag);

    const size_t lds_bytes = (size_t)ROWS_WG * SC_PITCH * 4      // score strip
                           + (size_t)ROWS_WG * BDm * 4           // su rows
                           + (size_t)BDm * Nn * 4                // sv (head)
                           + (size_t)ROWS_WG * 32 * 4            // adjacency bits
                           + (size_t)2 * 32 * FRAG * 2;          // bf16 A-frag buffer
    attn_kernel<<<dim3(Bb * Hh * (Nn / ROWS_WG)), dim3(64), lds_bytes, stream>>>(
        qfrag, kfrag, vfrag, adjbw, su, sv, adj_scale, thr, att, accum);

    out_proj_kernel<<<dim3(Bb * Nn), dim3(256), 0, stream>>>(
        att, x, w_o1, b_o1, w_o2, b_o2, out);

    finalize_kernel<<<dim3(1), dim3(1), 0, stream>>>(
        accum, l1_w, ent_w, out + (size_t)Bb * Nn * Cc);
}
